// Net_59021440582335
// MI455X (gfx1250) — matrix operand order repaired and checked by reading, not compiled, not run on hardware
//
#include <hip/hip_runtime.h>

#define N_NODES 100000
#define N_EDGES 1600000
#define F_IN    512
#define HID     128
#define NCLS    64

typedef __attribute__((ext_vector_type(16))) __bf16       v16bf;
typedef __attribute__((ext_vector_type(2)))  __bf16       v2bf;
typedef __attribute__((ext_vector_type(8)))  float        v8f;

struct Frag32 { uint4 lo, hi; };   // 32B: one WMMA 16-bf16 operand
static_assert(sizeof(Frag32) == 32, "frag size");

__device__ __forceinline__ unsigned short f2bf(float f) {
  unsigned int u = __builtin_bit_cast(unsigned int, f);
  u += 0x7FFFu + ((u >> 16) & 1u);   // round-to-nearest-even
  return (unsigned short)(u >> 16);
}

// Packed f32x2 -> bf16x2: lowers to v_cvt_pk_bf16_f32 on gfx1250 (verified).
__device__ __forceinline__ unsigned int pk2bf(float lo, float hi) {
  v2bf v;
  v[0] = (__bf16)lo;
  v[1] = (__bf16)hi;
  return __builtin_bit_cast(unsigned int, v);
}

__device__ __forceinline__ void atomAddF32(float* p, float v) {
  __hip_atomic_fetch_add(p, v, __ATOMIC_RELAXED, __HIP_MEMORY_SCOPE_AGENT);
}

// ---------------------------------------------------------------------------
// bf16 WMMA GEMM:  C[M x HOUT] = A[M x K] (f32) @ W[K x HOUT] + bias
// W pre-transposed bf16: Wt[col][k] so B fragments are contiguous.
// Block: 256 threads = 8 waves; block tile 128 rows x HOUT.
// A: straight to registers (each lane's fragment == its own 64B of A rows).
// B: double-buffered LDS (b128 in/out).
// K-loop unrolled by 2 so all register-array indices are compile-time
// (prevents scratch demotion of the A prefetch buffers).  T = K/32 is even.
// ---------------------------------------------------------------------------
template <int NT, bool RELU>
__global__ __launch_bounds__(256, 1)
void gemm_bf16_wmma(const float* __restrict__ A, int M, int K,
                    const unsigned short* __restrict__ Wt,  // [HOUT][K] bf16
                    const float* __restrict__ bias,
                    float* __restrict__ C) {
  constexpr int HOUT = NT * 16;
  constexpr int LSTR = 12;                       // dwords per lane slot (8 + 4 pad)
  __shared__ unsigned int Bs[2 * NT * 32 * LSTR];

  const int tid  = threadIdx.x;
  const int wave = tid >> 5;
  const int lane = tid & 31;
  const int n    = lane & 15;      // column within 16x16 tile
  const int hl   = lane >> 4;      // lane half (selects K groups)
  const int rowBase = blockIdx.x * 128;

  // ---- A source: this lane's own fragment rows/columns ----
  int ar = rowBase + wave * 16 + n;
  if (ar >= M) ar = M - 1;                       // clamp (stores predicated)
  const float* aRow = A + (size_t)ar * K + hl * 16;

  // ---- B staging source: slot == tid (for tid < NT*32) ----
  const int bslot = tid;                         // (t*32 + l)
  const int bcol  = (bslot >> 5) * 16 + (bslot & 15);
  const int bhl   = (bslot >> 4) & 1;
  const unsigned short* bSrc = Wt + (size_t)bcol * K + bhl * 16;

  const v8f vzero = {0.f, 0.f, 0.f, 0.f, 0.f, 0.f, 0.f, 0.f};
  v8f acc[NT];
#pragma unroll
  for (int t = 0; t < NT; ++t) acc[t] = vzero;

  float4 raE[4], raO[4];    // two named A-prefetch sets (no dynamic indexing)
  uint4  rb[2];

  auto loadA = [&](int k0, float4 (&r)[4]) {
    const float* p = aRow + k0;
    r[0] = *(const float4*)(p);
    r[1] = *(const float4*)(p + 4);
    r[2] = *(const float4*)(p + 8);
    r[3] = *(const float4*)(p + 12);
  };
  auto loadB = [&](int k0) {
    if constexpr (NT * 32 == 256) {
      const unsigned short* p = bSrc + k0;       // 16 consecutive bf16
      rb[0] = *(const uint4*)(p);
      rb[1] = *(const uint4*)(p + 8);
    } else {
      if (tid < NT * 32) {
        const unsigned short* p = bSrc + k0;
        rb[0] = *(const uint4*)(p);
        rb[1] = *(const uint4*)(p + 8);
      }
    }
  };
  auto storeB = [&](int buf) {
    if constexpr (NT * 32 == 256) {
      unsigned int* bs = Bs + buf * (NT * 32 * LSTR) + bslot * LSTR;
      *(uint4*)(bs)     = rb[0];
      *(uint4*)(bs + 4) = rb[1];
    } else {
      if (tid < NT * 32) {
        unsigned int* bs = Bs + buf * (NT * 32 * LSTR) + bslot * LSTR;
        *(uint4*)(bs)     = rb[0];
        *(uint4*)(bs + 4) = rb[1];
      }
    }
  };
  auto compute = [&](int buf, const float4 (&r)[4]) {
    // Build A fragment: dword j <-> K pair (16*hl+2j, +1), contiguous k order as in B
    Frag32 fa;
    fa.lo.x = pk2bf(r[0].x, r[0].y);
    fa.lo.y = pk2bf(r[0].z, r[0].w);
    fa.lo.z = pk2bf(r[1].x, r[1].y);
    fa.lo.w = pk2bf(r[1].z, r[1].w);
    fa.hi.x = pk2bf(r[2].x, r[2].y);
    fa.hi.y = pk2bf(r[2].z, r[2].w);
    fa.hi.z = pk2bf(r[3].x, r[3].y);
    fa.hi.w = pk2bf(r[3].z, r[3].w);
    v16bf a = __builtin_bit_cast(v16bf, fa);

    // Software-pipelined B fragments: load tile t+1 while WMMA t executes
    const unsigned int* bsBase = Bs + buf * (NT * 32 * LSTR) + lane * LSTR;
    Frag32 fb[2];
    fb[0].lo = *(const uint4*)(bsBase);           // ds_load_b128
    fb[0].hi = *(const uint4*)(bsBase + 4);
#pragma unroll
    for (int t = 0; t < NT; ++t) {
      if (t + 1 < NT) {
        const unsigned int* bs = bsBase + (t + 1) * (32 * LSTR);
        fb[(t + 1) & 1].lo = *(const uint4*)(bs);
        fb[(t + 1) & 1].hi = *(const uint4*)(bs + 4);
      }
      v16bf b = __builtin_bit_cast(v16bf, fb[t & 1]);
      acc[t] = __builtin_amdgcn_wmma_f32_16x16x32_bf16(
          false, a, false, b, (short)0, acc[t], false, false);
    }
  };

  const int T = K / 32;     // 16 or 4: always even
  loadA(0, raE);
  loadB(0);
  storeB(0);

  for (int kt = 0; kt < T; kt += 2) {
    // ---- even step: LDS buf 0, regs raE ----
    __syncthreads();
    loadB((kt + 1) * 32);                        // kt+1 <= T-1 always
    loadA((kt + 1) * 32, raO);
    storeB(1);
    compute(0, raE);
    // ---- odd step: LDS buf 1, regs raO ----
    __syncthreads();
    if (kt + 2 < T) {
      loadB((kt + 2) * 32);
      loadA((kt + 2) * 32, raE);
      storeB(0);
    }
    compute(1, raO);
  }

  // Epilogue: bias (+ReLU); C layout: VGPR j -> M=j (lanes 0-15) / M=j+8 (16-31)
#pragma unroll
  for (int t = 0; t < NT; ++t) {
    int col = t * 16 + n;
    float bv = bias[col];
#pragma unroll
    for (int j = 0; j < 8; ++j) {
      int row = rowBase + wave * 16 + hl * 8 + j;
      if (row < M) {
        float v = acc[t][j] + bv;
        if (RELU) v = fmaxf(v, 0.0f);
        C[(size_t)row * HOUT + col] = v;
      }
    }
  }
}

// ---------------------------------------------------------------------------
// Graph normalization + aggregation kernels (activations are L2-resident)
// ---------------------------------------------------------------------------
__global__ void init_deg(float* deg, int n) {
  int i = blockIdx.x * blockDim.x + threadIdx.x;
  if (i < n) deg[i] = 1.0f;  // self-loop
}

__global__ void accum_deg(const int* __restrict__ dst, float* deg, int e) {
  int i = blockIdx.x * blockDim.x + threadIdx.x;
  if (i < e) atomAddF32(&deg[dst[i]], 1.0f);
}

__global__ void make_dinv(float* d, int n) {
  int i = blockIdx.x * blockDim.x + threadIdx.x;
  if (i < n) d[i] = rsqrtf(d[i]);  // deg >= 1 always (self-loops)
}

// out[i,:] = h[i,:] * dinv[i]^2   (self-loop term; non-atomic init of out)
__global__ void selfloop_init(const float* __restrict__ h,
                              const float* __restrict__ dinv,
                              float* __restrict__ out, int n) {
  int t = blockIdx.x * blockDim.x + threadIdx.x;
  int node = t >> 5;
  int c = (t & 31) * 4;
  if (node < n) {
    float s = dinv[node]; s *= s;
    float4 v = *(const float4*)(h + (size_t)node * HID + c);
    v.x *= s; v.y *= s; v.z *= s; v.w *= s;
    *(float4*)(out + (size_t)node * HID + c) = v;
  }
}

// One wave per edge: out[dst,:] += h[src,:] * dinv[src]*dinv[dst]
__global__ void edge_scatter(const int* __restrict__ src,
                             const int* __restrict__ dst,
                             const float* __restrict__ dinv,
                             const float* __restrict__ h,
                             float* __restrict__ out, int e) {
  int wid  = (int)((blockIdx.x * (size_t)blockDim.x + threadIdx.x) >> 5);
  int lane = threadIdx.x & 31;
  if (wid < e) {
    int s = src[wid], d = dst[wid];
    float w = dinv[s] * dinv[d];
    float4 v = *(const float4*)(h + (size_t)s * HID + lane * 4);
    float* o = out + (size_t)d * HID + lane * 4;
    atomAddF32(o + 0, v.x * w);
    atomAddF32(o + 1, v.y * w);
    atomAddF32(o + 2, v.z * w);
    atomAddF32(o + 3, v.w * w);
  }
}

__global__ void relu_vec4(float* x, int n4) {
  int i = blockIdx.x * blockDim.x + threadIdx.x;
  if (i < n4) {
    float4 v = ((float4*)x)[i];
    v.x = fmaxf(v.x, 0.f); v.y = fmaxf(v.y, 0.f);
    v.z = fmaxf(v.z, 0.f); v.w = fmaxf(v.w, 0.f);
    ((float4*)x)[i] = v;
  }
}

// f32 W[K][HOUT] -> bf16 Wt[HOUT][K] (transpose so B fragments are contiguous)
__global__ void cvt_w_transpose(const float* __restrict__ in,
                                unsigned short* __restrict__ out,
                                int K, int HOUT) {
  int i = blockIdx.x * blockDim.x + threadIdx.x;
  if (i < K * HOUT) {
    int k = i / HOUT, c = i % HOUT;              // coalesced read
    out[(size_t)c * K + k] = f2bf(in[i]);
  }
}

// One wave per node, 64 classes -> 2 per lane; shuffle reductions.
__global__ void log_softmax64(const float* __restrict__ in,
                              float* __restrict__ out, int n) {
  int wid  = (int)((blockIdx.x * (size_t)blockDim.x + threadIdx.x) >> 5);
  int lane = threadIdx.x & 31;
  if (wid < n) {
    const float* row = in + (size_t)wid * NCLS;
    float a = row[lane], b = row[lane + 32];
    float m = fmaxf(a, b);
#pragma unroll
    for (int off = 16; off >= 1; off >>= 1) m = fmaxf(m, __shfl_xor(m, off, 32));
    float s = __expf(a - m) + __expf(b - m);
#pragma unroll
    for (int off = 16; off >= 1; off >>= 1) s += __shfl_xor(s, off, 32);
    float ls = m + __logf(s);
    float* orow = out + (size_t)wid * NCLS;
    orow[lane]      = a - ls;
    orow[lane + 32] = b - ls;
  }
}

// ---------------------------------------------------------------------------
static inline size_t alignup(size_t x) { return (x + 255) & ~(size_t)255; }

extern "C" void kernel_launch(void* const* d_in, const int* in_sizes, int n_in,
                              void* d_out, int out_size, void* d_ws, size_t ws_size,
                              hipStream_t stream) {
  const float* x    = (const float*)d_in[0];
  const int*   ei   = (const int*)d_in[1];       // [2,E]: row0=src, row1=dst
  const float* W0   = (const float*)d_in[2];
  const float* b0   = (const float*)d_in[3];
  const float* W1   = (const float*)d_in[4];
  const float* b1   = (const float*)d_in[5];
  const float* W2   = (const float*)d_in[6];
  const float* b2   = (const float*)d_in[7];
  const float* fc1w = (const float*)d_in[8];
  const float* fc1b = (const float*)d_in[9];
  const float* fc2w = (const float*)d_in[10];
  const float* fc2b = (const float*)d_in[11];
  const float* fc3w = (const float*)d_in[12];
  const float* fc3b = (const float*)d_in[13];
  float* out = (float*)d_out;

  const int* e_src = ei;
  const int* e_dst = ei + N_EDGES;

  // Workspace carve (~103 MB total)
  char* p = (char*)d_ws;
  float* dinv = (float*)p;          p += alignup((size_t)N_NODES * 4);
  float* hbuf = (float*)p;          p += alignup((size_t)N_NODES * HID * 4);
  float* abuf = (float*)p;          p += alignup((size_t)N_NODES * HID * 4);
  unsigned short* wbf0 = (unsigned short*)p; p += alignup((size_t)F_IN * HID * 2);
  unsigned short* wbf1 = (unsigned short*)p; p += alignup((size_t)HID * HID * 2);
  unsigned short* wbf2 = (unsigned short*)p; p += alignup((size_t)HID * HID * 2);
  unsigned short* wfc1 = (unsigned short*)p; p += alignup((size_t)HID * HID * 2);
  unsigned short* wfc2 = (unsigned short*)p; p += alignup((size_t)HID * HID * 2);
  unsigned short* wfc3 = (unsigned short*)p; p += alignup((size_t)HID * NCLS * 2);

  // Weight conversion f32 -> bf16, transposed (tiny)
  cvt_w_transpose<<<(F_IN * HID + 255) / 256, 256, 0, stream>>>(W0,   wbf0, F_IN, HID);
  cvt_w_transpose<<<(HID * HID + 255) / 256, 256, 0, stream>>>(W1,   wbf1, HID, HID);
  cvt_w_transpose<<<(HID * HID + 255) / 256, 256, 0, stream>>>(W2,   wbf2, HID, HID);
  cvt_w_transpose<<<(HID * HID + 255) / 256, 256, 0, stream>>>(fc1w, wfc1, HID, HID);
  cvt_w_transpose<<<(HID * HID + 255) / 256, 256, 0, stream>>>(fc2w, wfc2, HID, HID);
  cvt_w_transpose<<<(HID * NCLS + 255) / 256, 256, 0, stream>>>(fc3w, wfc3, HID, NCLS);

  // Symmetric GCN normalization: deg (with self-loops) -> dinv = deg^-1/2
  init_deg<<<(N_NODES + 255) / 256, 256, 0, stream>>>(dinv, N_NODES);
  accum_deg<<<(N_EDGES + 255) / 256, 256, 0, stream>>>(e_dst, dinv, N_EDGES);
  make_dinv<<<(N_NODES + 255) / 256, 256, 0, stream>>>(dinv, N_NODES);

  const int gemmBlocks = (N_NODES + 127) / 128;   // 782
  const int nodeWaves  = (N_NODES * 32 + 255) / 256;
  const int edgeWaves  = (int)(((size_t)N_EDGES * 32 + 255) / 256);
  const int reluBlocks = (N_NODES * HID / 4 + 255) / 256;

  // conv1: h = x @ W0 + b0 ; aggregate ; relu
  gemm_bf16_wmma<8, false><<<gemmBlocks, 256, 0, stream>>>(x, N_NODES, F_IN, wbf0, b0, hbuf);
  selfloop_init<<<nodeWaves, 256, 0, stream>>>(hbuf, dinv, abuf, N_NODES);
  edge_scatter<<<edgeWaves, 256, 0, stream>>>(e_src, e_dst, dinv, hbuf, abuf, N_EDGES);
  relu_vec4<<<reluBlocks, 256, 0, stream>>>(abuf, N_NODES * HID / 4);

  // conv2
  gemm_bf16_wmma<8, false><<<gemmBlocks, 256, 0, stream>>>(abuf, N_NODES, HID, wbf1, b1, hbuf);
  selfloop_init<<<nodeWaves, 256, 0, stream>>>(hbuf, dinv, abuf, N_NODES);
  edge_scatter<<<edgeWaves, 256, 0, stream>>>(e_src, e_dst, dinv, hbuf, abuf, N_EDGES);
  relu_vec4<<<reluBlocks, 256, 0, stream>>>(abuf, N_NODES * HID / 4);

  // conv3
  gemm_bf16_wmma<8, false><<<gemmBlocks, 256, 0, stream>>>(abuf, N_NODES, HID, wbf2, b2, hbuf);
  selfloop_init<<<nodeWaves, 256, 0, stream>>>(hbuf, dinv, abuf, N_NODES);
  edge_scatter<<<edgeWaves, 256, 0, stream>>>(e_src, e_dst, dinv, hbuf, abuf, N_EDGES);
  relu_vec4<<<reluBlocks, 256, 0, stream>>>(abuf, N_NODES * HID / 4);

  // fc1 (+relu fused), fc2 (+relu fused), fc3
  gemm_bf16_wmma<8, true ><<<gemmBlocks, 256, 0, stream>>>(abuf, N_NODES, HID, wfc1, fc1b, hbuf);
  gemm_bf16_wmma<8, true ><<<gemmBlocks, 256, 0, stream>>>(hbuf, N_NODES, HID, wfc2, fc2b, abuf);
  gemm_bf16_wmma<4, false><<<gemmBlocks, 256, 0, stream>>>(abuf, N_NODES, HID, wfc3, fc3b, hbuf);

  // log-softmax over 64 classes
  log_softmax64<<<nodeWaves, 256, 0, stream>>>(hbuf, out, N_NODES);
}
